// Attention_Block_9199819948230
// MI455X (gfx1250) — matrix operand order repaired and checked by reading, not compiled, not run on hardware
//
#include <hip/hip_runtime.h>
#include <cstdint>

#define BB 4
#define SS 1024
#define DD 1024
#define HH 16
#define DHD 64
#define KT 64   // k-tile rows per iteration

typedef __bf16 bf16;
typedef __attribute__((ext_vector_type(4)))  __bf16 bf16x4;
typedef __attribute__((ext_vector_type(16))) __bf16 v16bf;
typedef __attribute__((ext_vector_type(8)))  float  v8f;

// ---------------------------------------------------------------------------
// WMMA helpers (CDNA5 wave32, V_WMMA_F32_16X16X32_BF16)
// Fragment layouts per CDNA5 ISA 7.12.2:
//  A (16x32 bf16): lane 0-15 rows M, VGPR v holds K = (v<4?2v:16+2(v-4)) + 8*(lane>=16)
//  B (32x16 bf16): lane%16 = N; these helpers put K = 2v,2v+1 (+16 for lanes 16-31) in VGPR v for A and B
//  C/D (16x16 f32): VGPR r, lane l -> M = r + 8*(l>>4), N = l&15
// ---------------------------------------------------------------------------

__device__ __forceinline__ v8f wmma_bf16(v16bf a, v16bf b, v8f c) {
  return __builtin_amdgcn_wmma_f32_16x16x32_bf16(false, a, false, b,
                                                 (short)0, c, false, false);
}

__device__ __forceinline__ void wait_async0() {
#if __has_builtin(__builtin_amdgcn_s_wait_asynccnt)
  __builtin_amdgcn_s_wait_asynccnt(0);
#else
  asm volatile("s_wait_asynccnt 0x0" ::: "memory");
#endif
}

__device__ __forceinline__ bf16x4 cvt4(float4 f) {
  bf16x4 r;
  r[0] = (bf16)f.x; r[1] = (bf16)f.y; r[2] = (bf16)f.z; r[3] = (bf16)f.w;
  return r;
}

__device__ __forceinline__ v16bf load_A_frag(const bf16* p, int ld) {
  const int lane = threadIdx.x & 31;
  const int m = lane & 15;
  const int g = lane >> 4;
  v16bf a;
#pragma unroll
  for (int v = 0; v < 8; ++v) {
    const int kk = g * 16 + 2 * v;
    a[2 * v]     = p[m * ld + kk];
    a[2 * v + 1] = p[m * ld + kk + 1];
  }
  return a;
}

// B[k][n] sourced from row-major [n][k] bf16 storage (multiply by stored^T)
__device__ __forceinline__ v16bf load_B_fragT(const bf16* p, int ld) {
  const int lane = threadIdx.x & 31;
  const int n = lane & 15;
  const int g = lane >> 4;
  v16bf b;
#pragma unroll
  for (int v = 0; v < 8; ++v) {
    const int kk = g * 16 + 2 * v;
    b[2 * v]     = p[n * ld + kk];
    b[2 * v + 1] = p[n * ld + kk + 1];
  }
  return b;
}

// B[k][n] sourced from row-major [k][n] fp32 LDS storage (async-loaded V tile)
__device__ __forceinline__ v16bf load_B_frag_f32(const float* p, int ld) {
  const int lane = threadIdx.x & 31;
  const int n = lane & 15;
  const int g = lane >> 4;
  v16bf b;
#pragma unroll
  for (int v = 0; v < 8; ++v) {
    const int kk = g * 16 + 2 * v;
    b[2 * v]     = (bf16)p[kk * ld + n];
    b[2 * v + 1] = (bf16)p[(kk + 1) * ld + n];
  }
  return b;
}

// ---------------------------------------------------------------------------
// Kernel 1: fused flash attention.
//  - att_scores = QK^T/8 + c*scores - 1e8*(1-mask)  -> d_out (written once)
//  - online softmax, ctx = softmax(att_scores) @ V  -> workspace (B,S,D)
// Block: 256 threads (8 waves). Tile: 128 q-rows x 64 k-rows per step.
// V tile: GLOBAL_LOAD_ASYNC_TO_LDS_B128 (ASYNCcnt), overlapped with QK^T.
// Next iteration's K/V/scores pulled ahead with global_prefetch_b8.
// ---------------------------------------------------------------------------
__global__ __launch_bounds__(256) void flash_attn_kernel(
    const float* __restrict__ q, const float* __restrict__ k,
    const float* __restrict__ v, const float* __restrict__ mask,
    const float* __restrict__ scores, const float* __restrict__ cptr,
    float* __restrict__ out_scores, float* __restrict__ ctx) {
  constexpr int LQ = 72, LK = 72, LVF = 68, LP = 34;
  __shared__ bf16  sQ[128 * LQ];       // 18.0 KB
  __shared__ bf16  sK[KT * LK];        //  9.0 KB
  __shared__ float sVf[KT * LVF];      // 17.0 KB (fp32, async-filled)
  __shared__ bf16  sP[8][16 * LP];     //  8.5 KB per-wave P staging

  const int tid  = threadIdx.x;
  const int wid  = tid >> 5;
  const int lane = tid & 31;
  const int g    = lane >> 4;
  const int nn   = lane & 15;
  const int qt   = blockIdx.x;          // q tile (128 rows)
  const int bh   = blockIdx.y;          // b*H + h
  const int b    = bh / HH;
  const int h    = bh % HH;
  const float cval = cptr[0];

  // ---- load Q tile (128 x 64) as bf16, float4 loads + bf16x4 LDS stores ----
  {
    const float* qp = q + ((size_t)b * SS + (size_t)qt * 128) * DD + h * DHD;
#pragma unroll
    for (int i = 0; i < 8; ++i) {           // 2048 float4 slots
      const int linear = tid + i * 256;
      const int r  = linear >> 4;
      const int c4 = (linear & 15) << 2;
      const float4 f = *(const float4*)(qp + (size_t)r * DD + c4);
      *(bf16x4*)&sQ[r * LQ + c4] = cvt4(f);
    }
  }

  float mrow[8], lrow[8];
  v8f Oacc[4];
#pragma unroll
  for (int r = 0; r < 8; ++r) { mrow[r] = -3.0e38f; lrow[r] = 0.f; }
#pragma unroll
  for (int t = 0; t < 4; ++t) Oacc[t] = {};

  for (int kt = 0; kt < SS / KT; ++kt) {
    __syncthreads();  // previous iteration's sK/sVf readers are done

    // ---- async V tile: global fp32 -> LDS fp32, no VGPR round trip ----
    {
      const float* vp = v + ((size_t)b * SS + (size_t)kt * KT) * DD + h * DHD;
#pragma unroll
      for (int i = 0; i < 4; ++i) {
        const int linear = tid + i * 256;      // 0..1023 float4 slots
        const int r  = linear >> 4;
        const int c4 = (linear & 15) << 2;
        const float* gp = vp + (size_t)r * DD + c4;
        const unsigned ldsoff = (unsigned)(uintptr_t)(&sVf[r * LVF + c4]);
        asm volatile("global_load_async_to_lds_b128 %0, %1, off"
                     :: "v"(ldsoff), "v"(gp)
                     : "memory");
      }
    }

    // ---- prefetch next iteration's K/V tiles and scores chunk ----
    if (kt + 1 < SS / KT) {
      const size_t nt_base = ((size_t)b * SS + (size_t)(kt + 1) * KT) * DD + h * DHD;
      const int pr = tid >> 1, phalf = (tid & 1) * 32;   // 128 rows x 2 lines
      __builtin_prefetch(k + nt_base + (size_t)pr * DD + phalf, 0, 3);
      __builtin_prefetch(v + nt_base + (size_t)pr * DD + phalf, 0, 3);
      const int prow = qt * 128 + wid * 16 + nn;
      const size_t pidx = ((size_t)bh * SS + prow) * SS + (kt + 1) * KT + g * 32;
      __builtin_prefetch(&scores[pidx], 0, 3);
    }

    // ---- K tile (64 x 64): float4 loads + bf16x4 LDS stores ----
    {
      const float* kp = k + ((size_t)b * SS + (size_t)kt * KT) * DD + h * DHD;
#pragma unroll
      for (int i = 0; i < 4; ++i) {           // 1024 float4 slots
        const int linear = tid + i * 256;
        const int r  = linear >> 4;
        const int c4 = (linear & 15) << 2;
        const float4 f = *(const float4*)(kp + (size_t)r * DD + c4);
        *(bf16x4*)&sK[r * LK + c4] = cvt4(f);
      }
    }
    __syncthreads();  // sK ready (sVf not needed yet)

    // ---- score tiles: wave owns a 16-row strip, 4 column tiles ----
    v8f st[4];
#pragma unroll
    for (int nt = 0; nt < 4; ++nt) {
      v8f acc = {};
#pragma unroll
      for (int kk = 0; kk < 64; kk += 32) {
        v16bf a  = load_A_frag(&sQ[(wid * 16) * LQ + kk], LQ);
        v16bf bb = load_B_fragT(&sK[(nt * 16) * LK + kk], LK);
        acc = wmma_bf16(a, bb, acc);
      }
      const int kcol = kt * KT + nt * 16 + nn;
      const float mpen = -1e8f * (1.0f - mask[b * SS + kcol]);
#pragma unroll
      for (int r = 0; r < 8; ++r) {
        const int qrow = qt * 128 + wid * 16 + r + 8 * g;
        const size_t sidx = ((size_t)bh * SS + qrow) * SS + kcol;
        const float sc = acc[r] * 0.125f + cval * scores[sidx] + mpen;
        out_scores[sidx] = sc;
        acc[r] = sc;
      }
      st[nt] = acc;
    }

    // ---- online softmax row-stat update (half-wave owns rows r+8g) ----
    float alpha[8];
#pragma unroll
    for (int r = 0; r < 8; ++r) {
      float tm = st[0][r];
#pragma unroll
      for (int nt = 1; nt < 4; ++nt) tm = fmaxf(tm, st[nt][r]);
#pragma unroll
      for (int off = 1; off < 16; off <<= 1)
        tm = fmaxf(tm, __shfl_xor(tm, off, 16));
      const float mnew = fmaxf(mrow[r], tm);
      alpha[r] = __expf(mrow[r] - mnew);
      mrow[r]  = mnew;
      lrow[r] *= alpha[r];
    }
#pragma unroll
    for (int t = 0; t < 4; ++t)
#pragma unroll
      for (int r = 0; r < 8; ++r) Oacc[t][r] = Oacc[t][r] * alpha[r];

#pragma unroll
    for (int nt = 0; nt < 4; ++nt)
#pragma unroll
      for (int r = 0; r < 8; ++r) {
        const float p = __expf(st[nt][r] - mrow[r]);
        st[nt][r] = p;
        lrow[r] += p;
      }

    // ---- V tile must now be resident: drain ASYNCcnt, then barrier ----
    wait_async0();
    __syncthreads();

    // ---- P @ V : stage 16x32 P chunks per wave through LDS ----
    bf16* sPw = sP[wid];
#pragma unroll
    for (int kk2 = 0; kk2 < 2; ++kk2) {
#pragma unroll
      for (int t = 0; t < 2; ++t) {
        const int nt = kk2 * 2 + t;
#pragma unroll
        for (int r = 0; r < 8; ++r)
          sPw[(r + 8 * g) * LP + t * 16 + nn] = (bf16)st[nt][r];
      }
      v16bf a = load_A_frag(sPw, LP);  // same-wave LDS RAW: DScnt ordered
#pragma unroll
      for (int dt = 0; dt < 4; ++dt) {
        v16bf bb = load_B_frag_f32(&sVf[(kk2 * 32) * LVF + dt * 16], LVF);
        Oacc[dt] = wmma_bf16(a, bb, Oacc[dt]);
      }
    }
  }

  // ---- finalize: 1/rowsum and store ctx (fp32) ----
#pragma unroll
  for (int r = 0; r < 8; ++r) {
#pragma unroll
    for (int off = 1; off < 16; off <<= 1)
      lrow[r] += __shfl_xor(lrow[r], off, 16);
    lrow[r] = 1.0f / lrow[r];
  }
#pragma unroll
  for (int dt = 0; dt < 4; ++dt)
#pragma unroll
    for (int r = 0; r < 8; ++r) {
      const int srow = qt * 128 + wid * 16 + r + 8 * g;
      ctx[((size_t)b * SS + srow) * DD + h * DHD + dt * 16 + nn] =
          Oacc[dt][r] * lrow[r];
    }
}

// ---------------------------------------------------------------------------
// Kernel 2: generic two-source GEMM  C[m,n] = sum_k A1[m,k]*B1[n,k]
//                                          + sum_k A2[m,k]*B2[n,k]
// Block tile 128x128, 8 waves in 4(m) x 2(n); each wave: 2x4 16x16 tiles.
// ---------------------------------------------------------------------------
__global__ __launch_bounds__(256) void gemm2_kernel(
    const float* __restrict__ A1, int lda1,
    const float* __restrict__ B1, int ldb1, int K1,
    const float* __restrict__ A2, int lda2,
    const float* __restrict__ B2, int ldb2, int K2,
    float* __restrict__ C, int N) {
  constexpr int LA = 40, LB = 40;
  __shared__ bf16 sA[128 * LA];
  __shared__ bf16 sB[128 * LB];

  const int tid  = threadIdx.x;
  const int wid  = tid >> 5;
  const int lane = tid & 31;
  const int g    = lane >> 4;
  const int nn   = lane & 15;
  const int wq   = wid & 3;   // 32-row quadrant
  const int nh   = wid >> 2;  // 64-col half
  const int m0   = blockIdx.y * 128;
  const int n0   = blockIdx.x * 128;

  v8f acc[2][4];
#pragma unroll
  for (int i = 0; i < 2; ++i)
#pragma unroll
    for (int j = 0; j < 4; ++j) acc[i][j] = {};

  for (int src = 0; src < 2; ++src) {
    const float* A  = src ? A2 : A1;
    const float* Bw = src ? B2 : B1;
    const int lda = src ? lda2 : lda1;
    const int ldb = src ? ldb2 : ldb1;
    const int K   = src ? K2 : K1;
    if (A == nullptr) continue;
    for (int k0 = 0; k0 < K; k0 += 32) {
      __syncthreads();
      // stage 128x32 chunks of A and B: float4 loads, bf16x4 LDS stores
#pragma unroll
      for (int i = 0; i < 4; ++i) {          // 1024 float4 slots each
        const int linear = tid + i * 256;
        const int r  = linear >> 3;
        const int c4 = (linear & 7) << 2;
        const float4 fa = *(const float4*)(A  + (size_t)(m0 + r) * lda + k0 + c4);
        const float4 fb = *(const float4*)(Bw + (size_t)(n0 + r) * ldb + k0 + c4);
        *(bf16x4*)&sA[r * LA + c4] = cvt4(fa);
        *(bf16x4*)&sB[r * LB + c4] = cvt4(fb);
      }
      // prefetch next K-chunk (128B covers one row's 32 floats)
      if (k0 + 32 < K) {
        const int r = tid & 127;
        const float* pp = (tid < 128)
                              ? &A[(size_t)(m0 + r) * lda + k0 + 32]
                              : &Bw[(size_t)(n0 + r) * ldb + k0 + 32];
        __builtin_prefetch(pp, 0, 3);
      }
      __syncthreads();
#pragma unroll
      for (int mt = 0; mt < 2; ++mt) {
        v16bf a = load_A_frag(&sA[(wq * 32 + mt * 16) * LA], LA);
#pragma unroll
        for (int nt = 0; nt < 4; ++nt) {
          v16bf bb = load_B_fragT(&sB[(nh * 64 + nt * 16) * LB], LB);
          acc[mt][nt] = wmma_bf16(a, bb, acc[mt][nt]);
        }
      }
    }
  }

#pragma unroll
  for (int mt = 0; mt < 2; ++mt)
#pragma unroll
    for (int nt = 0; nt < 4; ++nt)
#pragma unroll
      for (int r = 0; r < 8; ++r) {
        const int row = m0 + wq * 32 + mt * 16 + r + 8 * g;
        const int col = n0 + nh * 64 + nt * 16 + nn;
        C[(size_t)row * N + col] = acc[mt][nt][r];
      }
}

// ---------------------------------------------------------------------------
// Kernel 3: rowwise LayerNorm over D=1024, one row per 256-thread block.
// ---------------------------------------------------------------------------
__global__ __launch_bounds__(256) void layernorm_kernel(
    const float* __restrict__ yraw, const float* __restrict__ w,
    const float* __restrict__ bsh, float* __restrict__ out) {
  __shared__ float red0[8];
  __shared__ float red1[8];
  const int row = blockIdx.x;
  const int tid = threadIdx.x;
  const float* yp = yraw + (size_t)row * DD;

  float vals[4];
  float s = 0.f, s2 = 0.f;
#pragma unroll
  for (int i = 0; i < 4; ++i) {
    vals[i] = yp[tid + i * 256];
    s  += vals[i];
    s2 += vals[i] * vals[i];
  }
#pragma unroll
  for (int off = 1; off < 32; off <<= 1) {
    s  += __shfl_xor(s, off, 32);
    s2 += __shfl_xor(s2, off, 32);
  }
  const int wid = tid >> 5, lane = tid & 31;
  if (lane == 0) { red0[wid] = s; red1[wid] = s2; }
  __syncthreads();
  if (wid == 0) {
    float a  = (lane < 8) ? red0[lane] : 0.f;
    float b2 = (lane < 8) ? red1[lane] : 0.f;
#pragma unroll
    for (int off = 1; off < 8; off <<= 1) {
      a  += __shfl_xor(a, off, 32);
      b2 += __shfl_xor(b2, off, 32);
    }
    if (lane == 0) { red0[0] = a; red1[0] = b2; }
  }
  __syncthreads();
  const float mean = red0[0] * (1.0f / DD);
  const float var  = red1[0] * (1.0f / DD) - mean * mean;
  const float rstd = rsqrtf(var + 1e-5f);
  float* op = out + (size_t)row * DD;
#pragma unroll
  for (int i = 0; i < 4; ++i) {
    const int cc = tid + i * 256;
    op[cc] = (vals[i] - mean) * rstd * w[cc] + bsh[cc];
  }
}

// ---------------------------------------------------------------------------
extern "C" void kernel_launch(void* const* d_in, const int* in_sizes, int n_in,
                              void* d_out, int out_size, void* d_ws,
                              size_t ws_size, hipStream_t stream) {
  const float* q       = (const float*)d_in[0];
  const float* k       = (const float*)d_in[1];
  const float* v       = (const float*)d_in[2];
  const float* mask    = (const float*)d_in[3];
  const float* scores  = (const float*)d_in[4];
  const float* W_proj  = (const float*)d_in[5];
  const float* W_minus = (const float*)d_in[6];
  const float* ln_w    = (const float*)d_in[7];
  const float* ln_b    = (const float*)d_in[8];
  const float* c       = (const float*)d_in[9];

  float* out_y      = (float*)d_out;
  float* out_scores = out_y + (size_t)BB * SS * DD;

  float* ctx  = (float*)d_ws;                      // (B,S,D)
  float* xbuf = ctx + (size_t)BB * SS * DD;        // (B,S,D)
  float* yraw = xbuf + (size_t)BB * SS * DD;       // (B,S,D)

  // 1) fused attention: att_scores out + ctx = softmax @ V
  flash_attn_kernel<<<dim3(SS / 128, BB * HH), 256, 0, stream>>>(
      q, k, v, mask, scores, c, out_scores, ctx);

  // 2) x = ctx @ W_proj^T
  gemm2_kernel<<<dim3(DD / 128, (BB * SS) / 128), 256, 0, stream>>>(
      ctx, DD, W_proj, DD, DD, nullptr, 0, nullptr, 0, 0, xbuf, DD);

  // 3) y_raw = q @ W_minus[:, :D]^T + x @ W_minus[:, D:]^T
  gemm2_kernel<<<dim3(DD / 128, (BB * SS) / 128), 256, 0, stream>>>(
      q, DD, W_minus, 2 * DD, DD, xbuf, DD, W_minus + DD, 2 * DD, DD, yraw, DD);

  // 4) LayerNorm -> y
  layernorm_kernel<<<BB * SS, 256, 0, stream>>>(yraw, ln_w, ln_b, out_y);
}
